// OBELISK_60842506715608
// MI455X (gfx1250) — matrix operand order repaired and checked by reading, not compiled, not run on hardware
//
#include <hip/hip_runtime.h>

// ---------------- CDNA5 WMMA types ----------------
typedef __bf16 bf16;
typedef __attribute__((ext_vector_type(8)))  __bf16 v8bf;
typedef __attribute__((ext_vector_type(16))) __bf16 v16bf;
typedef __attribute__((ext_vector_type(8)))  float  v8f;

// ---------------- problem constants ----------------
#define HIN 192
#define WIN 160
#define DIN 256
#define PD 96
#define PH 80
#define PW 128            // avgpool out (z,y,x)
#define CD 48
#define CHH 40
#define CW 64             // conv0 out spatial
#define M0 (CD*CHH*CW)    // 122880
#define OM 64
#define ON 53
#define OO 85
#define NPT (OM*ON*OO)    // 288320 output points
#define KOFF 48
#define K1 192            // conv1 K (=4*48)
#define C96 96
#define MT (NPT/16)       // 18020 row tiles (exact)
#define PZ 66
#define PY 55
#define PX 87
#define NPAD (PZ*PY*PX)   // 315810 padded points

// ---------------- stage 1: avg_pool3d k3 s2 p1 (/27) ----------------
__global__ void k_avgpool(const float* __restrict__ in, float* __restrict__ out) {
  int t = blockIdx.x * 256 + threadIdx.x;
  if (t >= PD * PH * PW) return;
  int x = t % PW, y = (t / PW) % PH, z = t / (PW * PH);
  float s = 0.f;
  for (int dz = 0; dz < 3; ++dz) { int iz = 2 * z - 1 + dz; if ((unsigned)iz >= HIN) continue;
    for (int dy = 0; dy < 3; ++dy) { int iy = 2 * y - 1 + dy; if ((unsigned)iy >= WIN) continue;
      for (int dx = 0; dx < 3; ++dx) { int ix = 2 * x - 1 + dx; if ((unsigned)ix >= DIN) continue;
        s += in[((long)iz * WIN + iy) * DIN + ix]; } } }
  out[t] = s * (1.f / 27.f);
}

// ---------------- stage 2: conv 5^3, 1->4, stride 2, pad 2 ----------------
__global__ void k_conv0(const float* __restrict__ P, const float* __restrict__ w0,
                        float* __restrict__ raw) {
  int t = blockIdx.x * 256 + threadIdx.x;
  if (t >= 4 * M0) return;
  int o = t / M0, m = t % M0;
  int x = m % CW, y = (m / CW) % CHH, z = m / (CW * CHH);
  float s = 0.f;
  for (int kz = 0; kz < 5; ++kz) { int iz = 2 * z + kz - 2; if ((unsigned)iz >= PD) continue;
    for (int ky = 0; ky < 5; ++ky) { int iy = 2 * y + ky - 2; if ((unsigned)iy >= PH) continue;
      for (int kx = 0; kx < 5; ++kx) { int ix = 2 * x + kx - 2; if ((unsigned)ix >= PW) continue;
        s += P[((long)iz * PH + iy) * PW + ix] * w0[((o * 5 + kz) * 5 + ky) * 5 + kx]; } } }
  raw[(long)m * 4 + o] = s;   // [m][c] interleaved for BN stats
}

// ---------------- BN statistics (deterministic 2-pass) ----------------
__global__ void k_bn_partial(const float* __restrict__ y, float* __restrict__ part,
                             int M, int C, int parts, int rowsPerBlock) {
  int t = threadIdx.x;
  int c = t % C, p = t / C;
  long row0 = (long)blockIdx.x * rowsPerBlock;
  long rend = row0 + rowsPerBlock; if (rend > M) rend = M;
  float s = 0.f, sq = 0.f;
  for (long r = row0 + p; r < rend; r += parts) {
    float v = y[r * C + c];
    s += v; sq += v * v;
  }
  long base = ((long)blockIdx.x * parts + p) * 2 * C;
  part[base + c] = s;
  part[base + C + c] = sq;
}

__global__ void k_bn_final(const float* __restrict__ part, int nchunks, int C, float invM,
                           const float* __restrict__ g, const float* __restrict__ b,
                           float* __restrict__ ss) {
  int c = threadIdx.x;
  if (c >= C) return;
  float s = 0.f, sq = 0.f;
  for (int i = 0; i < nchunks; ++i) {           // fixed order -> deterministic
    s  += part[(long)i * 2 * C + c];
    sq += part[(long)i * 2 * C + C + c];
  }
  float mu  = s * invM;
  float var = sq * invM - mu * mu;
  float sc  = g[c] * rsqrtf(var + 1e-5f);
  ss[c]     = sc;
  ss[C + c] = b[c] - mu * sc;
}

// ---------------- BN apply for conv0 -> img [c][m] f32 ----------------
__global__ void k_apply0(const float* __restrict__ raw, const float* __restrict__ ss,
                         float* __restrict__ img) {
  int t = blockIdx.x * 256 + threadIdx.x;
  if (t >= 4 * M0) return;
  int m = t / 4, c = t % 4;
  float v = raw[t] * ss[c] + ss[4 + c];
  img[(long)c * M0 + m] = v > 0.f ? v : 0.f;
}

// ---------------- weight repacks (f32 -> bf16, WMMA-friendly N x K) ----------
__global__ void k_pack_cast(const float* __restrict__ w, bf16* __restrict__ o, int n) {
  int t = blockIdx.x * 256 + threadIdx.x;
  if (t < n) o[t] = (bf16)w[t];
}
__global__ void k_pack_w2(const float* __restrict__ w2, bf16* __restrict__ o) {
  int t = blockIdx.x * 256 + threadIdx.x;
  if (t >= 96 * 96 * 27) return;
  int j = t / (96 * 27); int rem = t % (96 * 27);
  int i = rem / 27; int tap = rem % 27;
  o[(long)j * 2592 + tap * 96 + i] = (bf16)w2[t];   // K order: tap-major, channel inner
}
__global__ void k_pack_w3(const float* __restrict__ w3, bf16* __restrict__ o) {
  int t = blockIdx.x * 256 + threadIdx.x;
  if (t >= 32 * 96) return;
  int j = t / 96, i = t % 96;
  o[t] = (j < 24) ? (bf16)w3[j * 96 + i] : (bf16)0.f;  // pad N 24 -> 32
}

// ---------------- stage 3: trilinear sampling (align_corners=True, zero pad) ----
__global__ void k_sample(const float* __restrict__ img, const float* __restrict__ offs,
                         bf16* __restrict__ S) {
  long t = (long)blockIdx.x * 256 + threadIdx.x;
  if (t >= (long)NPT * KOFF) return;
  int k = (int)(t % KOFF);
  int n = (int)(t / KOFF);
  int xi = n % OO, yi = (n / OO) % ON, zi = n / (OO * ON);
  float bx = (2.f * xi + 1.f) / OO - 1.f;
  float by = (2.f * yi + 1.f) / ON - 1.f;
  float bz = (2.f * zi + 1.f) / OM - 1.f;
  float a0 = 0.f, a1 = 0.f, a2 = 0.f, a3 = 0.f;
  #pragma unroll
  for (int set = 0; set < 2; ++set) {
    float sgn = set ? -1.f : 1.f;
    const float* o = offs + ((long)set * KOFF + k) * 3;
    float ix = (bx + o[0] + 1.f) * (0.5f * (CW  - 1));
    float iy = (by + o[1] + 1.f) * (0.5f * (CHH - 1));
    float iz = (bz + o[2] + 1.f) * (0.5f * (CD  - 1));
    float x0 = floorf(ix), y0 = floorf(iy), z0 = floorf(iz);
    #pragma unroll
    for (int dz = 0; dz < 2; ++dz)
      #pragma unroll
      for (int dy = 0; dy < 2; ++dy)
        #pragma unroll
        for (int dx = 0; dx < 2; ++dx) {
          float xf = x0 + dx, yf = y0 + dy, zf = z0 + dz;
          float w = (1.f - fabsf(ix - xf)) * (1.f - fabsf(iy - yf)) * (1.f - fabsf(iz - zf));
          bool valid = (xf >= 0.f) && (xf < (float)CW) && (yf >= 0.f) && (yf < (float)CHH)
                    && (zf >= 0.f) && (zf < (float)CD);
          if (valid && w > 0.f) {
            int xc = (int)xf, yc = (int)yf, zc = (int)zf;
            long base = ((long)zc * CHH + yc) * CW + xc;
            float wv = sgn * w;
            a0 += wv * img[base];
            a1 += wv * img[(long)M0 + base];
            a2 += wv * img[2L * M0 + base];
            a3 += wv * img[3L * M0 + base];
          }
        }
  }
  bf16* row = S + (long)n * K1 + k;   // channel q = c*48 + k
  row[0]   = (bf16)a0;
  row[48]  = (bf16)a1;
  row[96]  = (bf16)a2;
  row[144] = (bf16)a3;
}

// ---------------- WMMA helpers ----------------
__device__ __forceinline__ v16bf load_a_frag(const bf16* row, int g) {
  // lane m = L%16 holds A[m][kbase + 16g + 0..15]
  v8bf lo = *(const v8bf*)(row + 16 * g);
  v8bf hi = *(const v8bf*)(row + 16 * g + 8);
  return __builtin_shufflevector(lo, hi, 0,1,2,3,4,5,6,7,8,9,10,11,12,13,14,15);
}

// ---------------- GEMM1: Y[n][96] = S[n][192] * W1^T (bf16 WMMA) -------------
__global__ void __launch_bounds__(256) k_gemm1(const bf16* __restrict__ A,
                                               const bf16* __restrict__ W,
                                               float* __restrict__ Y) {
  int wave = threadIdx.x >> 5, lane = threadIdx.x & 31;
  int mt = blockIdx.x * 8 + wave;
  if (mt >= MT) return;
  int mr = lane & 15, g = lane >> 4;
  const bf16* arow = A + (long)(mt * 16 + mr) * K1;
  v8f zero = {0.f,0.f,0.f,0.f,0.f,0.f,0.f,0.f};
  v8f acc[6];
  #pragma unroll
  for (int j = 0; j < 6; ++j) acc[j] = zero;
  #pragma unroll
  for (int ks = 0; ks < 6; ++ks) {
    v16bf af = load_a_frag(arow + ks * 32, g);
    #pragma unroll
    for (int jt = 0; jt < 6; ++jt) {
      v16bf bfr = *(const v16bf*)(W + (long)(jt * 16 + mr) * K1 + ks * 32 + 16 * g);
      acc[jt] = __builtin_amdgcn_wmma_f32_16x16x32_bf16(false, af, false, bfr,
                                                        (short)0, acc[jt], false, false);
    }
  }
  #pragma unroll
  for (int jt = 0; jt < 6; ++jt)
    #pragma unroll
    for (int r = 0; r < 8; ++r)
      Y[(long)(mt * 16 + r + 8 * g) * C96 + jt * 16 + mr] = acc[jt][r];
}

// ---------------- BN apply -> padded channels-last bf16 x1 ----------------
__global__ void k_apply1(const float* __restrict__ Yv, const float* __restrict__ ss,
                         bf16* __restrict__ Xp) {
  long t = (long)blockIdx.x * 256 + threadIdx.x;
  if (t >= (long)NPT * C96) return;
  int c = (int)(t % C96);
  int n = (int)(t / C96);
  float v = Yv[t] * ss[c] + ss[C96 + c];
  v = v > 0.f ? v : 0.f;
  int x = n % OO, y = (n / OO) % ON, z = n / (OO * ON);
  long p = (((long)(z + 1) * PY + (y + 1)) * PX + (x + 1)) * C96 + c;
  Xp[p] = (bf16)v;
}

__global__ void k_zero32(unsigned int* __restrict__ p, long n) {
  long stride = (long)gridDim.x * 256;
  for (long i = (long)blockIdx.x * 256 + threadIdx.x; i < n; i += stride) p[i] = 0u;
}

// ---------------- GEMM2: 3x3x3 conv as im2col bf16 WMMA (K = 27*96) ----------
// Weights for the current tap (96x96 bf16 = 18KB) are staged in LDS,
// double-buffered and software-pipelined: B reads come from the DS pipe
// (ds_load_b128) instead of VMEM, shared by all 8 waves of the block.
#define TAPW64 2304      // 18432 B per tap as 64-bit words
#define CPT 9            // 2304 / 256 threads

__global__ void __launch_bounds__(256) k_gemm2(const bf16* __restrict__ Xp,
                                               const bf16* __restrict__ W,
                                               float* __restrict__ Y) {
  __shared__ unsigned long long lw[2][TAPW64];

  int tid  = threadIdx.x;
  int wave = tid >> 5, lane = tid & 31;
  int mt   = blockIdx.x * 8 + wave;
  bool active = (mt < MT);
  int mtc  = active ? mt : (MT - 1);       // keep all waves alive for barriers
  int mr   = lane & 15, g = lane >> 4;

  int n = mtc * 16 + mr;
  int x = n % OO, y = (n / OO) % ON, z = n / (OO * ON);
  long p0 = ((long)z * PY + y) * PX + x;   // padded-origin (taps 0..2 cover -1..+1)

  v8f zero = {0.f,0.f,0.f,0.f,0.f,0.f,0.f,0.f};
  v8f acc[6];
  #pragma unroll
  for (int j = 0; j < 6; ++j) acc[j] = zero;

  // per-thread slice of the tap weight block: CPT 8-byte words
  unsigned long long reg[CPT];
  #pragma unroll
  for (int q = 0; q < CPT; ++q) {          // prologue: issue loads for tap 0
    int idx = tid + q * 256;
    int row = idx / 24, off = idx % 24;    // 24 x u64 per 96-channel row
    reg[q] = *((const unsigned long long*)(W + (long)row * 2592) + off);
  }

  for (int tap = 0; tap < 27; ++tap) {
    int buf = tap & 1;
    // commit staged weights to LDS (compiler waits loadcnt before ds_store)
    #pragma unroll
    for (int q = 0; q < CPT; ++q) lw[buf][tid + q * 256] = reg[q];
    __syncthreads();

    // issue next tap's weight loads (no wait; overlaps with WMMA below)
    if (tap < 26) {
      #pragma unroll
      for (int q = 0; q < CPT; ++q) {
        int idx = tid + q * 256;
        int row = idx / 24, off = idx % 24;
        reg[q] = *((const unsigned long long*)(W + (long)row * 2592 + (tap + 1) * 96) + off);
      }
    }

    // compute this tap: A from global (192B/row), B from LDS
    int tz = tap / 9, ty = (tap / 3) % 3, tx = tap % 3;
    const bf16* ap = Xp + (p0 + ((long)tz * PY + ty) * PX + tx) * C96;
    __builtin_prefetch(ap + C96, 0, 1);    // next-tap activation row
    const bf16* lb = (const bf16*)&lw[buf][0];
    #pragma unroll
    for (int cc = 0; cc < 3; ++cc) {
      v16bf af = load_a_frag(ap + cc * 32, g);
      #pragma unroll
      for (int jt = 0; jt < 6; ++jt) {
        v16bf bfr = *(const v16bf*)(lb + (jt * 16 + mr) * C96 + cc * 32 + 16 * g);
        acc[jt] = __builtin_amdgcn_wmma_f32_16x16x32_bf16(false, af, false, bfr,
                                                          (short)0, acc[jt], false, false);
      }
    }
    __syncthreads();   // all waves done reading lw[buf] before it is rewritten
  }

  if (active) {
    #pragma unroll
    for (int jt = 0; jt < 6; ++jt)
      #pragma unroll
      for (int r = 0; r < 8; ++r)
        Y[(long)(mt * 16 + r + 8 * g) * C96 + jt * 16 + mr] = acc[jt][r];
  }
}

// ---------------- BN apply -> x2 bf16 row-major [n][96] ----------------
__global__ void k_apply2(const float* __restrict__ Yv, const float* __restrict__ ss,
                         bf16* __restrict__ X2) {
  long t = (long)blockIdx.x * 256 + threadIdx.x;
  if (t >= (long)NPT * C96) return;
  int c = (int)(t % C96);
  float v = Yv[t] * ss[c] + ss[C96 + c];
  X2[t] = (bf16)(v > 0.f ? v : 0.f);
}

// ---------------- GEMM3: 1x1 conv 96->24 + bias, channel-major output --------
__global__ void __launch_bounds__(256) k_gemm3(const bf16* __restrict__ A,
                                               const bf16* __restrict__ W,
                                               const float* __restrict__ b3,
                                               float* __restrict__ out) {
  int wave = threadIdx.x >> 5, lane = threadIdx.x & 31;
  int mt = blockIdx.x * 8 + wave;
  if (mt >= MT) return;
  int mr = lane & 15, g = lane >> 4;
  const bf16* arow = A + (long)(mt * 16 + mr) * C96;
  v8f zero = {0.f,0.f,0.f,0.f,0.f,0.f,0.f,0.f};
  v8f acc[2] = {zero, zero};
  #pragma unroll
  for (int ks = 0; ks < 3; ++ks) {
    v16bf af = load_a_frag(arow + ks * 32, g);
    #pragma unroll
    for (int jt = 0; jt < 2; ++jt) {
      v16bf bfr = *(const v16bf*)(W + (long)(jt * 16 + mr) * C96 + ks * 32 + 16 * g);
      acc[jt] = __builtin_amdgcn_wmma_f32_16x16x32_bf16(false, af, false, bfr,
                                                        (short)0, acc[jt], false, false);
    }
  }
  #pragma unroll
  for (int jt = 0; jt < 2; ++jt) {
    int j = jt * 16 + mr;
    if (j < 24) {
      float bb = b3[j];
      #pragma unroll
      for (int r = 0; r < 8; ++r)
        out[(long)j * NPT + mt * 16 + r + 8 * g] = acc[jt][r] + bb;
    }
  }
}

// ---------------- launch ----------------
extern "C" void kernel_launch(void* const* d_in, const int* in_sizes, int n_in,
                              void* d_out, int out_size, void* d_ws, size_t ws_size,
                              hipStream_t stream) {
  const float* in_img = (const float*)d_in[0];
  const float* offs   = (const float*)d_in[1];
  const float* w0     = (const float*)d_in[2];
  const float* g0     = (const float*)d_in[3];
  const float* b0     = (const float*)d_in[4];
  const float* w1     = (const float*)d_in[5];
  const float* g1     = (const float*)d_in[6];
  const float* b1     = (const float*)d_in[7];
  const float* w2     = (const float*)d_in[8];
  const float* g2     = (const float*)d_in[9];
  const float* b2     = (const float*)d_in[10];
  const float* w3     = (const float*)d_in[11];
  const float* b3     = (const float*)d_in[12];
  float* out = (float*)d_out;

  char* ws = (char*)d_ws;
  float* P      = (float*)(ws + 0);              //  3,932,160 B
  float* c0raw  = (float*)(ws + 3932160);        //  1,966,080 B
  float* img    = (float*)(ws + 5898240);        //  1,966,080 B
  float* ssbuf  = (float*)(ws + 7864320);        //      1,024 B (scale|shift)
  bf16*  w1b    = (bf16*) (ws + 7865344);        //     36,864 B
  bf16*  w2b    = (bf16*) (ws + 7902208);        //    497,664 B
  bf16*  w3b    = (bf16*) (ws + 8399872);        //      6,144 B
  float* part   = (float*)(ws + 8406016);        //    870,400 B (BN partials)
  bf16*  Smp    = (bf16*) (ws + 9276416);        // 110,714,880 B (sampled, bf16 [n][192])
  float* Yb     = (float*)(ws + 119991296);      // 110,714,880 B (raw GEMM out f32 [n][96])
  bf16*  X1p    = (bf16*) (ws + 230706176);      //  60,635,520 B (padded ch-last bf16)
  bf16*  X2     = (bf16*) (ws + 9276416);        // alias Smp (dead after gemm1)

  // stage 1+2: pool, conv0, BN0, ReLU
  k_avgpool<<<(PD * PH * PW + 255) / 256, 256, 0, stream>>>(in_img, P);
  k_conv0<<<(4 * M0 + 255) / 256, 256, 0, stream>>>(P, w0, c0raw);
  k_bn_partial<<<240, 128, 0, stream>>>(c0raw, part, M0, 4, 32, 512);
  k_bn_final<<<1, 128, 0, stream>>>(part, 240 * 32, 4, 1.f / (float)M0, g0, b0, ssbuf);
  k_apply0<<<(4 * M0 + 255) / 256, 256, 0, stream>>>(c0raw, ssbuf, img);

  // weight repacks
  k_pack_cast<<<(18432 + 255) / 256, 256, 0, stream>>>(w1, w1b, 18432);
  k_pack_w2<<<(248832 + 255) / 256, 256, 0, stream>>>(w2, w2b);
  k_pack_w3<<<(3072 + 255) / 256, 256, 0, stream>>>(w3, w3b);

  // stage 3: sampling -> bf16 [n][192]
  k_sample<<<(int)(((long)NPT * KOFF + 255) / 256), 256, 0, stream>>>(img, offs, Smp);

  // stage 4: 1x1 conv (WMMA) + BN1 + ReLU -> padded ch-last bf16
  k_gemm1<<<(MT + 7) / 8, 256, 0, stream>>>(Smp, w1b, Yb);
  k_bn_partial<<<564, 192, 0, stream>>>(Yb, part, NPT, 96, 2, 512);
  k_bn_final<<<1, 128, 0, stream>>>(part, 564 * 2, 96, 1.f / (float)NPT, g1, b1, ssbuf);
  k_zero32<<<2048, 256, 0, stream>>>((unsigned int*)X1p, (long)NPAD * C96 / 2);
  k_apply1<<<(int)(((long)NPT * C96 + 255) / 256), 256, 0, stream>>>(Yb, ssbuf, X1p);

  // stage 5: 3x3x3 conv (im2col WMMA, LDS-staged weights) + BN2 + ReLU
  k_gemm2<<<(MT + 7) / 8, 256, 0, stream>>>(X1p, w2b, Yb);
  k_bn_partial<<<564, 192, 0, stream>>>(Yb, part, NPT, 96, 2, 512);
  k_bn_final<<<1, 128, 0, stream>>>(part, 564 * 2, 96, 1.f / (float)NPT, g2, b2, ssbuf);
  k_apply2<<<(int)(((long)NPT * C96 + 255) / 256), 256, 0, stream>>>(Yb, ssbuf, X2);

  // stage 6: 1x1 conv 96->24 + bias, writes (24,64,53,85) channel-major
  k_gemm3<<<(MT + 7) / 8, 256, 0, stream>>>(X2, w3b, b3, out);
}
